// AdaptiveSignatureHedger_80874234184107
// MI455X (gfx1250) — matrix operand order repaired and checked by reading, not compiled, not run on hardware
//
#include <hip/hip_runtime.h>

// ---------------------------------------------------------------------------
// AdaptiveSignatureHedger — CDNA5 (gfx1250) implementation.
// B=32, T=2048, d=5, h=64, W=10.  wave32, WMMA bf16 16x16x32, f32 accumulate.
// Round 2: software-pipelined K-loops (2-deep) + dual accumulator chains so
// WMMAs wait on dscnt<=2 instead of draining counters every iteration.
// ---------------------------------------------------------------------------

#define B_DIM   32
#define T_DIM   2048
#define D_DIM   5
#define H_DIM   64
#define W_WIN   10
#define TILE_M  16
#define TILES_PER_ROW (T_DIM / TILE_M)   // 128

typedef __attribute__((ext_vector_type(16))) __bf16 v16bf;
typedef __attribute__((ext_vector_type(8)))  __bf16 v8bf;
typedef __attribute__((ext_vector_type(8)))  float  v8f;

static __device__ __forceinline__ __bf16 tobf(float x) { return (__bf16)x; }
static __device__ __forceinline__ float  frbf(__bf16 x) { return (float)x; }

// Packed-weight fragment table (each fragment = 32 lanes * 16 bf16 = 512 bf16)
// K padded to multiples of 32 with ZERO weight rows, so the A side can read a
// single shared signature vector without masking.
#define FRAG_P1   0     // K=5   ->  1 ktile * 4 ntiles
#define FRAG_P2   4     // K=30  ->  1 * 4
#define FRAG_P3   8     // K=155 ->  5 * 4
#define FRAG_P4   28    // K=780 -> 25 * 4
#define FRAG_W1   128   // K=69  ->  3 * 4
#define FRAG_W2   140   // K=64, N=32 -> 2 * 2
#define FRAG_TOTAL 144

static __device__ __forceinline__ v8f wmma_bf16(v16bf a, v16bf b, v8f c) {
  return __builtin_amdgcn_wmma_f32_16x16x32_bf16(false, a, false, b,
                                                 (short)0, c, false, false);
}

// A-fragment (16x32 bf16) from a row-major LDS row. lane 0-15: M=lane,K kk..kk+7
// and 16+kk..16+kk+7 ; lanes 16-31 same M set, kk = 8.
static __device__ __forceinline__ v16bf load_a(const __bf16* row, int kt, int lane) {
  const int kk = (lane >> 4) * 8;
  v8bf lo = *(const v8bf*)(row + kt * 32 + kk);
  v8bf hi = *(const v8bf*)(row + kt * 32 + 16 + kk);
  v16bf a;
#pragma unroll
  for (int i = 0; i < 8; ++i) { a[i] = lo[i]; a[i + 8] = hi[i]; }
  return a;
}

static __device__ __forceinline__ v16bf load_b(const __bf16* wpack, int frag, int lane) {
  return *(const v16bf*)(wpack + frag * 512 + lane * 16);
}

// Software-pipelined GEMM over NKT ktiles: loads run 2 tiles ahead of the
// consuming WMMA, and even/odd ktiles accumulate into independent chains.
template <int NKT, int NTS>
static __device__ __forceinline__ v8f gemm_pipe(const __bf16* xrow,
                                                const __bf16* wpack,
                                                int fbase, int nt, int lane) {
  v8f c0 = {0.f, 0.f, 0.f, 0.f, 0.f, 0.f, 0.f, 0.f};
  v8f c1 = {0.f, 0.f, 0.f, 0.f, 0.f, 0.f, 0.f, 0.f};
  v16bf a0, b0, a1, b1;
  a0 = load_a(xrow, 0, lane);
  b0 = load_b(wpack, fbase + nt, lane);
  if (NKT > 1) {
    a1 = load_a(xrow, 1, lane);
    b1 = load_b(wpack, fbase + NTS + nt, lane);
  }
#pragma unroll
  for (int kt = 0; kt < NKT; ++kt) {
    if ((kt & 1) == 0) {
      c0 = wmma_bf16(a0, b0, c0);
      if (kt + 2 < NKT) {
        a0 = load_a(xrow, kt + 2, lane);
        b0 = load_b(wpack, fbase + (kt + 2) * NTS + nt, lane);
      }
    } else {
      c1 = wmma_bf16(a1, b1, c1);
      if (kt + 2 < NKT) {
        a1 = load_a(xrow, kt + 2, lane);
        b1 = load_b(wpack, fbase + (kt + 2) * NTS + nt, lane);
      }
    }
  }
  return (NKT > 1) ? (c0 + c1) : c0;
}

// ---------------------------------------------------------------------------
// Kernel 1: prelude — ret, vol (prefix scan), H-MLP, softmax gate.
// One block per batch row, 256 threads.  Writes mask-folded gate to ws.
// ---------------------------------------------------------------------------
__global__ __launch_bounds__(256) void prelude_kernel(
    const float* __restrict__ feat,
    const float* __restrict__ hw1, const float* __restrict__ hb1,
    const float* __restrict__ hw2, const float* __restrict__ hb2,
    const float* __restrict__ hw3, const float* __restrict__ hb3,
    const float* __restrict__ gw1, const float* __restrict__ gb1,
    const float* __restrict__ gw2, const float* __restrict__ gb2,
    float* __restrict__ wsGate) {
  __shared__ float retL[T_DIM];
  __shared__ float volL[T_DIM];
  __shared__ float chunkL[256];
  __shared__ float swh1[320], sbh1[32], swh2[1024], sbh2[32], swh3[32];
  __shared__ float sgw1[64], sgb1[32], sgw2[128], sgb2[4], sbh3[1];

  const int b   = blockIdx.x;
  const int tid = threadIdx.x;
  const float* fb = feat + (size_t)b * T_DIM * D_DIM;

  // returns of the price channel
  for (int t = tid; t < T_DIM; t += 256)
    retL[t] = (t == 0) ? 0.0f : (fb[t * D_DIM] - fb[(t - 1) * D_DIM]);

  // stage the small MLP weights
  for (int i = tid; i < 320;  i += 256) swh1[i] = hw1[i];
  for (int i = tid; i < 32;   i += 256) sbh1[i] = hb1[i];
  for (int i = tid; i < 1024; i += 256) swh2[i] = hw2[i];
  for (int i = tid; i < 32;   i += 256) sbh2[i] = hb2[i];
  for (int i = tid; i < 32;   i += 256) swh3[i] = hw3[i];
  for (int i = tid; i < 64;   i += 256) sgw1[i] = gw1[i];
  for (int i = tid; i < 32;   i += 256) sgb1[i] = gb1[i];
  for (int i = tid; i < 128;  i += 256) sgw2[i] = gw2[i];
  for (int i = tid; i < 4;    i += 256) sgb2[i] = gb2[i];
  if (tid == 0) sbh3[0] = hb3[0];
  __syncthreads();

  // inclusive prefix scan of |ret| -> vol
  {
    float run = 0.0f;
    const int base = tid * 8;
#pragma unroll
    for (int u = 0; u < 8; ++u) { run += fabsf(retL[base + u]); volL[base + u] = run; }
    chunkL[tid] = run;
  }
  __syncthreads();
  if (tid == 0) {
    float run = 0.0f;
    for (int j = 0; j < 256; ++j) { float v = chunkL[j]; chunkL[j] = run; run += v; }
  }
  __syncthreads();
  {
    const int base = tid * 8;
    const float off = chunkL[tid];
#pragma unroll
    for (int u = 0; u < 8; ++u) {
      int t = base + u;
      volL[t] = (volL[t] + off) / ((float)(t + 1) + 1e-8f);
    }
  }
  __syncthreads();

  // H-MLP + gate for 8 tokens per thread
  for (int u = 0; u < 8; ++u) {
    const int t = tid * 8 + u;
    const int base = (t < W_WIN) ? 0 : (t - W_WIN);
    float win[W_WIN];
#pragma unroll
    for (int i = 0; i < W_WIN; ++i) win[i] = retL[base + i];

    float h1v[32];
#pragma unroll
    for (int o = 0; o < 32; ++o) {
      float s = sbh1[o];
#pragma unroll
      for (int i = 0; i < W_WIN; ++i) s += win[i] * swh1[i * 32 + o];
      h1v[o] = fmaxf(s, 0.0f);
    }
    float h2v[32];
#pragma unroll
    for (int o = 0; o < 32; ++o) {
      float s = sbh2[o];
#pragma unroll
      for (int i = 0; i < 32; ++i) s += h1v[i] * swh2[i * 32 + o];
      h2v[o] = fmaxf(s, 0.0f);
    }
    float uacc = sbh3[0];
#pragma unroll
    for (int i = 0; i < 32; ++i) uacc += h2v[i] * swh3[i];
    const float Hv  = 0.5f / (1.0f + expf(-uacc));
    const float vol = volL[t];

    float gr[32];
#pragma unroll
    for (int o = 0; o < 32; ++o)
      gr[o] = fmaxf(Hv * sgw1[o] + vol * sgw1[32 + o] + sgb1[o], 0.0f);
    float lg[4];
#pragma unroll
    for (int k = 0; k < 4; ++k) {
      float s = sgb2[k];
#pragma unroll
      for (int i = 0; i < 32; ++i) s += gr[i] * sgw2[i * 4 + k];
      lg[k] = s;
    }
    float mx = fmaxf(fmaxf(lg[0], lg[1]), fmaxf(lg[2], lg[3]));
    float e0 = expf(lg[0] - mx), e1 = expf(lg[1] - mx);
    float e2 = expf(lg[2] - mx), e3 = expf(lg[3] - mx);
    const float inv = 1.0f / (e0 + e1 + e2 + e3);
    float* g = wsGate + ((size_t)b * T_DIM + t) * 4;
    g[0] = e0 * inv;                               // proj1 always live
    g[1] = (t >= 1) ? e1 * inv : 0.0f;             // fold masks m2..m4 here
    g[2] = (t >= 2) ? e2 * inv : 0.0f;
    g[3] = (t >= 3) ? e3 * inv : 0.0f;
  }
}

// ---------------------------------------------------------------------------
// Kernel 2: pack fp32 weights into bf16 WMMA B-fragment layout (zero-padded K)
// ---------------------------------------------------------------------------
__global__ __launch_bounds__(256) void pack_weights_kernel(
    const float* __restrict__ p1, const float* __restrict__ p2,
    const float* __restrict__ p3, const float* __restrict__ p4,
    const float* __restrict__ w1, const float* __restrict__ w2,
    __bf16* __restrict__ wpack) {
  const int tid = blockIdx.x * blockDim.x + threadIdx.x;
  if (tid >= FRAG_TOTAL * 32) return;
  const int fid  = tid >> 5;
  const int lane = tid & 31;

  const float* src; int Krows, Ncols, ntiles, fbase;
  if      (fid < FRAG_P2) { src = p1; Krows = 5;   Ncols = 64; ntiles = 4; fbase = FRAG_P1; }
  else if (fid < FRAG_P3) { src = p2; Krows = 30;  Ncols = 64; ntiles = 4; fbase = FRAG_P2; }
  else if (fid < FRAG_P4) { src = p3; Krows = 155; Ncols = 64; ntiles = 4; fbase = FRAG_P3; }
  else if (fid < FRAG_W1) { src = p4; Krows = 780; Ncols = 64; ntiles = 4; fbase = FRAG_P4; }
  else if (fid < FRAG_W2) { src = w1; Krows = 69;  Ncols = 64; ntiles = 4; fbase = FRAG_W1; }
  else                    { src = w2; Krows = 64;  Ncols = 32; ntiles = 2; fbase = FRAG_W2; }

  const int local = fid - fbase;
  const int kt = local / ntiles;
  const int nt = local % ntiles;
  const int n  = nt * 16 + (lane & 15);
  __bf16* dst = wpack + (size_t)fid * 512 + lane * 16;
#pragma unroll
  for (int e = 0; e < 16; ++e) {
    const int k = kt * 32 + (lane >> 4) * 8 + (e & 7) + (e >> 3) * 16;
    const float v = (k < Krows) ? src[k * Ncols + n] : 0.0f;
    dst[e] = tobf(v);
  }
}

// ---------------------------------------------------------------------------
// Kernel 3: main — signature build + gated projections + head, all via WMMA.
// One wave (32 threads) per 16-token tile.
// ---------------------------------------------------------------------------
__global__ __launch_bounds__(32) void hedger_main_kernel(
    const float* __restrict__ feat,
    const float* __restrict__ p1b, const float* __restrict__ p2b,
    const float* __restrict__ p3b, const float* __restrict__ p4b,
    const float* __restrict__ b1,  const float* __restrict__ gamma,
    const float* __restrict__ beta,const float* __restrict__ rmean,
    const float* __restrict__ rvar,const float* __restrict__ b2,
    const float* __restrict__ w3,  const float* __restrict__ b3,
    const float* __restrict__ wsGate, const __bf16* __restrict__ wpack,
    float* __restrict__ out) {
  __shared__ __align__(16) __bf16 xbuf[TILE_M][800];   // signature A matrix
  __shared__ __align__(16) __bf16 combA[TILE_M][96];   // [sig_repr | feat | 0] A
  __shared__ __align__(16) __bf16 h1buf[TILE_M][64];   // post-BN activations A
  __shared__ float sfeat[26][D_DIM];
  __shared__ float gm[TILE_M][4];
  __shared__ float red[TILE_M];

  const int b    = blockIdx.x / TILES_PER_ROW;
  const int t0   = (blockIdx.x % TILES_PER_ROW) * TILE_M;
  const int lane = threadIdx.x;
  const int m    = lane & 15;
  const int half8 = (lane >> 4) * 8;
  const float* fb = feat + (size_t)b * T_DIM * D_DIM;

  __builtin_prefetch((const void*)(wpack + FRAG_P4 * 512), 0, 1);

  // ---- phase 1: stage features window, gates (mask-folded), zero reduction
  for (int i = lane; i < 26 * D_DIM; i += 32) {
    const int rr = i / D_DIM, c = i % D_DIM;
    const int tg = t0 - W_WIN + rr;
    sfeat[rr][c] = (tg >= 0) ? fb[tg * D_DIM + c] : 0.0f;
  }
  if (lane < 16) {
    const float* g = wsGate + ((size_t)b * T_DIM + (t0 + m)) * 4;
#pragma unroll
    for (int k = 0; k < 4; ++k) gm[m][k] = g[k];
    red[m] = 0.0f;
  }
  __syncthreads();

  // ---- phase 2: signature levels 1-3 (suffix-sum recurrence), lanes 0-15
  if (lane < 16) {
    float S[D_DIM];
    float l2f[25];
    float RA[25];
    float l3f[125];
#pragma unroll
    for (int i = 0; i < D_DIM; ++i) S[i] = 0.0f;
#pragma unroll
    for (int i = 0; i < 25; ++i) { l2f[i] = 0.0f; RA[i] = 0.0f; }
#pragma unroll
    for (int i = 0; i < 125; ++i) l3f[i] = 0.0f;

#pragma unroll
    for (int r = W_WIN - 1; r >= 0; --r) {
      const bool valid = (t0 + m - W_WIN + r) >= 0;
      float dw[D_DIM];
#pragma unroll
      for (int i = 0; i < D_DIM; ++i)
        dw[i] = valid ? (sfeat[m + r + 1][i] - sfeat[m + r][i]) : 0.0f;
      // l3 contribution uses RA[r] = suffix over r' > r  (before adding A[r])
#pragma unroll
      for (int i = 0; i < D_DIM; ++i)
#pragma unroll
        for (int ef = 0; ef < 25; ++ef)
          l3f[i * 25 + ef] = fmaf(dw[i], RA[ef], l3f[i * 25 + ef]);
      // S becomes suffix-from-r ; A[r] = dw (x) S
#pragma unroll
      for (int j = 0; j < D_DIM; ++j) S[j] += dw[j];
#pragma unroll
      for (int i = 0; i < D_DIM; ++i)
#pragma unroll
        for (int j = 0; j < D_DIM; ++j) {
          const float a = dw[i] * S[j];
          l2f[i * D_DIM + j] += a;
          RA[i * D_DIM + j]  += a;
        }
    }
    // write x[0:155] and zero pad [780:800]
#pragma unroll
    for (int i = 0; i < D_DIM; ++i) xbuf[m][i] = tobf(S[i]);
#pragma unroll
    for (int i = 0; i < 25; ++i) xbuf[m][5 + i] = tobf(l2f[i]);
#pragma unroll
    for (int i = 0; i < 125; ++i) xbuf[m][30 + i] = tobf(l3f[i]);
#pragma unroll
    for (int i = 780; i < 800; ++i) xbuf[m][i] = tobf(0.0f);
  }
  __syncthreads();

  // ---- phase 3: level-4 = outer(l2f, l2f), split across the two lane halves
  {
    float l2v[25];
#pragma unroll
    for (int v = 0; v < 25; ++v) l2v[v] = frbf(xbuf[m][5 + v]);
    const int u0 = (lane < 16) ? 0 : 13;
    const int u1 = (lane < 16) ? 13 : 25;
    for (int u = u0; u < u1; ++u) {
      const float au = l2v[u];
#pragma unroll
      for (int v = 0; v < 25; ++v)
        xbuf[m][155 + u * 25 + v] = tobf(au * l2v[v]);
    }
  }
  __syncthreads();

  // ---- phase 4: gated signature projections via WMMA (4 N-tiles of 16)
  const __bf16* xrow = &xbuf[m][0];
#pragma unroll
  for (int nt = 0; nt < 4; ++nt) {
    const int ncol = nt * 16 + m;
    v8f sig = {0.f, 0.f, 0.f, 0.f, 0.f, 0.f, 0.f, 0.f};

    // proj4 : K = 780 (25 ktiles, pipelined, dual accumulators)
    {
      v8f c = gemm_pipe<25, 4>(xrow, wpack, FRAG_P4, nt, lane);
      const float bk = p4b[ncol];
#pragma unroll
      for (int r = 0; r < 8; ++r) sig[r] += gm[half8 + r][3] * (c[r] + bk);
    }
    // proj3 : K = 155 (5 ktiles)
    {
      v8f c = gemm_pipe<5, 4>(xrow, wpack, FRAG_P3, nt, lane);
      const float bk = p3b[ncol];
#pragma unroll
      for (int r = 0; r < 8; ++r) sig[r] += gm[half8 + r][2] * (c[r] + bk);
    }
    // proj2 : K = 30 (1 ktile, zero-padded weights)
    {
      v8f c = gemm_pipe<1, 4>(xrow, wpack, FRAG_P2, nt, lane);
      const float bk = p2b[ncol];
#pragma unroll
      for (int r = 0; r < 8; ++r) sig[r] += gm[half8 + r][1] * (c[r] + bk);
    }
    // proj1 : K = 5 (1 ktile, zero-padded weights)
    {
      v8f c = gemm_pipe<1, 4>(xrow, wpack, FRAG_P1, nt, lane);
      const float bk = p1b[ncol];
#pragma unroll
      for (int r = 0; r < 8; ++r) sig[r] += gm[half8 + r][0] * (c[r] + bk);
    }
    // D-layout -> combined A (bf16)
#pragma unroll
    for (int r = 0; r < 8; ++r) combA[half8 + r][ncol] = tobf(sig[r]);
  }
  // append raw features + zero pad to K=96
  if (lane < 16) {
#pragma unroll
    for (int c = 0; c < D_DIM; ++c) combA[m][64 + c] = tobf(sfeat[m + W_WIN][c]);
#pragma unroll
    for (int c = 69; c < 96; ++c) combA[m][c] = tobf(0.0f);
  }
  __syncthreads();

  // ---- phase 5: head layer 1 (69->64) + fused BatchNorm
  const __bf16* arow = &combA[m][0];
#pragma unroll
  for (int nt = 0; nt < 4; ++nt) {
    const int ncol = nt * 16 + m;
    v8f c = gemm_pipe<3, 4>(arow, wpack, FRAG_W1, nt, lane);
    const float b1c = b1[ncol];
    const float sc  = gamma[ncol] * rsqrtf(rvar[ncol] + 1e-5f);
    const float sh  = beta[ncol] - sc * rmean[ncol];
#pragma unroll
    for (int r = 0; r < 8; ++r) {
      const float h1v = fmaxf(c[r] + b1c, 0.0f);
      h1buf[half8 + r][ncol] = tobf(sc * h1v + sh);
    }
  }
  __syncthreads();

  // ---- phase 6: head layer 2 (64->32) + w3 dot via LDS float atomics
  const __bf16* hrow = &h1buf[m][0];
#pragma unroll
  for (int nt = 0; nt < 2; ++nt) {
    const int ncol = nt * 16 + m;
    v8f c = gemm_pipe<2, 2>(hrow, wpack, FRAG_W2, nt, lane);
    const float b2c = b2[ncol];
    const float w3c = w3[ncol];
#pragma unroll
    for (int r = 0; r < 8; ++r) {
      const float hv = fmaxf(c[r] + b2c, 0.0f);
      atomicAdd(&red[half8 + r], hv * w3c);
    }
  }
  __syncthreads();

  if (lane < 16)
    out[(size_t)b * T_DIM + (t0 + m)] = 1.5f * tanhf(red[m] + b3[0]);
}

// ---------------------------------------------------------------------------
extern "C" void kernel_launch(void* const* d_in, const int* in_sizes, int n_in,
                              void* d_out, int out_size, void* d_ws, size_t ws_size,
                              hipStream_t stream) {
  const float* feat = (const float*)d_in[0];
  const float* hw1  = (const float*)d_in[1];
  const float* hb1  = (const float*)d_in[2];
  const float* hw2  = (const float*)d_in[3];
  const float* hb2  = (const float*)d_in[4];
  const float* hw3  = (const float*)d_in[5];
  const float* hb3  = (const float*)d_in[6];
  const float* gw1  = (const float*)d_in[7];
  const float* gb1  = (const float*)d_in[8];
  const float* gw2  = (const float*)d_in[9];
  const float* gb2  = (const float*)d_in[10];
  const float* p1   = (const float*)d_in[11];
  const float* p1b  = (const float*)d_in[12];
  const float* p2   = (const float*)d_in[13];
  const float* p2b  = (const float*)d_in[14];
  const float* p3   = (const float*)d_in[15];
  const float* p3b  = (const float*)d_in[16];
  const float* p4   = (const float*)d_in[17];
  const float* p4b  = (const float*)d_in[18];
  const float* w1   = (const float*)d_in[19];
  const float* b1   = (const float*)d_in[20];
  const float* gamma= (const float*)d_in[21];
  const float* beta = (const float*)d_in[22];
  const float* rmean= (const float*)d_in[23];
  const float* rvar = (const float*)d_in[24];
  const float* w2   = (const float*)d_in[25];
  const float* b2   = (const float*)d_in[26];
  const float* w3   = (const float*)d_in[27];
  const float* b3   = (const float*)d_in[28];
  float* outp = (float*)d_out;

  float*  wsGate = (float*)d_ws;                                     // B*T*4 f32
  __bf16* wpack  = (__bf16*)((char*)d_ws + (size_t)B_DIM * T_DIM * 4 * sizeof(float));

  prelude_kernel<<<B_DIM, 256, 0, stream>>>(feat, hw1, hb1, hw2, hb2, hw3, hb3,
                                            gw1, gb1, gw2, gb2, wsGate);
  pack_weights_kernel<<<(FRAG_TOTAL * 32 + 255) / 256, 256, 0, stream>>>(
      p1, p2, p3, p4, w1, w2, wpack);
  hedger_main_kernel<<<B_DIM * TILES_PER_ROW, 32, 0, stream>>>(
      feat, p1b, p2b, p3b, p4b, b1, gamma, beta, rmean, rvar, b2, w3, b3,
      wsGate, wpack, outp);
}
